// RNNWavefunction_18511309045771
// MI455X (gfx1250) — matrix operand order repaired and checked by reading, not compiled, not run on hardware
//
#include <hip/hip_runtime.h>
#include <hip/hip_bf16.h>

#define HDIM  512
#define SEQ   512
#define BATCH 1024

typedef __attribute__((ext_vector_type(16))) __bf16 v16bf;
typedef __attribute__((ext_vector_type(8)))  __bf16 v8bf;
typedef __attribute__((ext_vector_type(8)))  float  v8f;

union Frag { v16bf v; v8bf h[2]; };

#define BM 128
#define BN 64
#define BK 32
#define PAD 8
#define NCHUNK (HDIM / BK)   // 16

__device__ __forceinline__ float sigmoidf_(float x) { return 1.0f / (1.0f + __expf(-x)); }

// One GEMM accumulation pass over K = HDIM with double-buffered LDS staging.
// GN is the accumulator group receiving the n-gate product (2 = inn from Wi,
// 3 = hn from Wh).  All accumulator indices are compile-time constants, so the
// compiler assigns fixed VGPR ranges to each WMMA (no cndmask selection).
template <int GN>
__device__ __forceinline__ void run_pass(
    const __bf16* __restrict__ Acur, const __bf16* __restrict__ Wcur,
    int m0, int n0, int tid, int waveM, int waveN, int l16, int hh,
    __bf16 (&sA)[2][BM][BK + PAD], __bf16 (&sB)[2][3][BN][BK + PAD],
    v8f (&acc)[4][2][2])
{
    // Per-chunk staging: each thread moves 2 A-vectors + 3 B-vectors (16B each).
    v8bf aReg[2], bReg[3];

    auto gload = [&](int kk) {
#pragma unroll
        for (int u = 0; u < 2; ++u) {
            const int idx = tid + u * 256;
            const int row = idx >> 2, col = (idx & 3) * 8;
            aReg[u] = *(const v8bf*)&Acur[(size_t)(m0 + row) * HDIM + kk + col];
        }
#pragma unroll
        for (int u = 0; u < 3; ++u) {
            const int idx = tid + u * 256;
            const int g = idx >> 8, rem = idx & 255;
            const int row = rem >> 2, col = (rem & 3) * 8;
            bReg[u] = *(const v8bf*)&Wcur[(size_t)g * HDIM * HDIM +
                                          (size_t)(n0 + row) * HDIM + kk + col];
        }
    };
    auto sstore = [&](int buf) {
#pragma unroll
        for (int u = 0; u < 2; ++u) {
            const int idx = tid + u * 256;
            const int row = idx >> 2, col = (idx & 3) * 8;
            *(v8bf*)&sA[buf][row][col] = aReg[u];
        }
#pragma unroll
        for (int u = 0; u < 3; ++u) {
            const int idx = tid + u * 256;
            const int g = idx >> 8, rem = idx & 255;
            const int row = rem >> 2, col = (rem & 3) * 8;
            *(v8bf*)&sB[buf][g][row][col] = bReg[u];
        }
    };

    gload(0);
    sstore(0);
    __syncthreads();

    for (int c = 0; c < NCHUNK; ++c) {
        const int  buf  = c & 1;
        const int  kk   = c * BK;
        const bool more = (c + 1) < NCHUNK;
        if (more) {
            gload(kk + BK);
            // prefetch chunk-after-next into the cache hierarchy
            if (c + 2 < NCHUNK)
                __builtin_prefetch(&Acur[(size_t)(m0 + (tid >> 2)) * HDIM + kk + 2 * BK], 0, 0);
        }

        // A fragments (ISA 16-bit A 16x32 layout): lane l (m=l&15, h=l>>4):
        //   elems 0..7  <- K = h*8 .. h*8+7
        //   elems 8..15 <- K = 16+h*8 .. 16+h*8+7
        Frag aF[2];
#pragma unroll
        for (int i = 0; i < 2; ++i) {
            const __bf16* pr = &sA[buf][waveM * 32 + i * 16 + l16][0];
            aF[i].h[0] = *(const v8bf*)(pr + hh * 8);
            aF[i].h[1] = *(const v8bf*)(pr + 16 + hh * 8);
        }
        // B fragments (ISA 16-bit B 32x16 layout): lane l (n=l&15, h=l>>4):
        //   elems 0..7 <- K = h*8 .. h*8+7, elems 8..15 <- K = 16+h*8 .. 16+h*8+7
#pragma unroll
        for (int j = 0; j < 2; ++j) {
#pragma unroll
            for (int g = 0; g < 3; ++g) {
                Frag bF;
                const __bf16* pb = &sB[buf][g][waveN * 32 + j * 16 + l16][0];
                bF.h[0] = *(const v8bf*)(pb + hh * 8);
                bF.h[1] = *(const v8bf*)(pb + 16 + hh * 8);
                constexpr int GRPOF[3] = {0, 1, GN};
                const int grp = GRPOF[g];   // g unrolled -> compile-time constant
#pragma unroll
                for (int i = 0; i < 2; ++i) {
                    acc[grp][i][j] = __builtin_amdgcn_wmma_f32_16x16x32_bf16(
                        false, aF[i].v, false, bF.v, (short)0, acc[grp][i][j],
                        false, false);
                }
            }
        }

        if (more) sstore(buf ^ 1);
        __syncthreads();
    }
}

// Fused GRU layer step:  G = A1@W1 (+ A2@W2 when SECOND) ; gates -> new hidden.
// !SECOND (layer 0): gh is rank-1 = xp*v0, carry = xp*Wemb + bemb.
//  SECOND (layer 1): gh = A2@W2, carry[] read directly.
template <bool SECOND>
__global__ __launch_bounds__(256, 1)
void gru_layer_kernel(const __bf16* __restrict__ A1, const __bf16* __restrict__ W1,
                      const __bf16* __restrict__ A2, const __bf16* __restrict__ W2,
                      const float* __restrict__ bias, const float* __restrict__ v0,
                      const int* __restrict__ xTok, int t,
                      const float* __restrict__ Wemb, const float* __restrict__ bemb,
                      const float* __restrict__ carry,
                      float* __restrict__ Hf, __bf16* __restrict__ Hb)
{
    __shared__ __bf16 sA[2][BM][BK + PAD];       // double-buffered activation tiles
    __shared__ __bf16 sB[2][3][BN][BK + PAD];    // double-buffered weight tiles

    const int tid   = threadIdx.x;
    const int lane  = tid & 31;
    const int wave  = tid >> 5;
    const int waveM = wave & 3;                  // 4 waves along M
    const int waveN = wave >> 2;                 // 2 waves along N
    const int m0    = blockIdx.y * BM;
    const int n0    = blockIdx.x * BN;
    const int l16   = lane & 15;
    const int hh    = lane >> 4;

    // groups: 0 = r (gi+gh), 1 = z (gi+gh), 2 = inn (gi n-gate), 3 = hn (gh n-gate)
    v8f acc[4][2][2];
#pragma unroll
    for (int g = 0; g < 4; ++g)
#pragma unroll
        for (int i = 0; i < 2; ++i)
#pragma unroll
            for (int j = 0; j < 2; ++j) acc[g][i][j] = v8f{};

    run_pass<2>(A1, W1, m0, n0, tid, waveM, waveN, l16, hh, sA, sB, acc);
    if constexpr (SECOND)
        run_pass<3>(A2, W2, m0, n0, tid, waveM, waveN, l16, hh, sA, sB, acc);

    // GRU epilogue.  C/D layout: VGPR r, lane l -> (M = r + (l>>4)*8, N = l&15).
#pragma unroll
    for (int i = 0; i < 2; ++i) {
        const int mg = m0 + waveM * 32 + i * 16 + hh * 8;
#pragma unroll
        for (int j = 0; j < 2; ++j) {
            const int ng = n0 + waveN * 32 + j * 16 + l16;
            const float br = bias[ng];
            const float bz = bias[HDIM + ng];
            const float bn = bias[2 * HDIM + ng];
            float v0r = 0.f, v0z = 0.f, v0n = 0.f, we = 0.f, be = 0.f;
            if constexpr (!SECOND) {
                v0r = v0[ng]; v0z = v0[HDIM + ng]; v0n = v0[2 * HDIM + ng];
                we = Wemb[ng]; be = bemb[ng];
            }
#pragma unroll
            for (int r = 0; r < 8; ++r) {
                const int m = mg + r;
                float rg, zg, hn, c;
                if constexpr (!SECOND) {
                    float xp = 0.f;
                    if (t > 0) xp = (float)xTok[(size_t)m * SEQ + (t - 1)];
                    rg = sigmoidf_(acc[0][i][j][r] + br + xp * v0r);
                    zg = sigmoidf_(acc[1][i][j][r] + bz + xp * v0z);
                    hn = xp * v0n;
                    c  = xp * we + be;
                } else {
                    rg = sigmoidf_(acc[0][i][j][r] + br);
                    zg = sigmoidf_(acc[1][i][j][r] + bz);
                    hn = acc[3][i][j][r];
                    c  = carry[(size_t)m * HDIM + ng];
                }
                const float inn = acc[2][i][j][r] + bn;
                const float nt  = tanhf(inn + rg * hn);
                const float o   = (1.f - zg) * nt + zg * c;
                Hf[(size_t)m * HDIM + ng] = o;
                Hb[(size_t)m * HDIM + ng] = (__bf16)o;
            }
        }
    }
}

// W (H x 3H f32, row-major) -> WT (3 x H x H bf16): WT[g][n][k] = W[k][g*H+n]
__global__ void transpose_w_kernel(const float* __restrict__ W, __bf16* __restrict__ WT)
{
    int idx = blockIdx.x * blockDim.x + threadIdx.x;
    int g   = idx / (HDIM * HDIM);
    int rem = idx % (HDIM * HDIM);
    int n   = rem / HDIM;
    int k   = rem % HDIM;
    WT[idx] = (__bf16)W[(size_t)k * 3 * HDIM + g * HDIM + n];
}

// v0[j] = sum_k Wemb[k] * Wh0[k][j]   (rank-1 collapse of layer-0's gh GEMM)
__global__ void v0_kernel(const float* __restrict__ Wemb, const float* __restrict__ Wh0,
                          float* __restrict__ v0)
{
    int j = blockIdx.x * blockDim.x + threadIdx.x;
    if (j >= 3 * HDIM) return;
    float s = 0.f;
    for (int k = 0; k < HDIM; ++k) s = fmaf(Wemb[k], Wh0[(size_t)k * 3 * HDIM + j], s);
    v0[j] = s;
}

// logits = h @ Wo + bo ; log_amp += log_softmax(logits)[x[:,t]]
__global__ void head_kernel(const float* __restrict__ h, const float* __restrict__ Wo,
                            const float* __restrict__ bo, const int* __restrict__ xTok,
                            int t, float* __restrict__ logamp)
{
    int b = blockIdx.x * blockDim.x + threadIdx.x;
    if (b >= BATCH) return;
    float l0 = bo[0], l1 = bo[1];
    const float* hr = h + (size_t)b * HDIM;
    for (int k = 0; k < HDIM; ++k) {
        float hv = hr[k];
        l0 = fmaf(hv, Wo[2 * k], l0);
        l1 = fmaf(hv, Wo[2 * k + 1], l1);
    }
    float mx  = fmaxf(l0, l1);
    float lse = mx + __logf(__expf(l0 - mx) + __expf(l1 - mx));
    int   xc  = xTok[(size_t)b * SEQ + t];
    logamp[b] += ((xc == 0) ? l0 : l1) - lse;
}

__global__ void finalize_kernel(const float* __restrict__ logamp, float* __restrict__ out,
                                int out_size)
{
    int b = blockIdx.x * blockDim.x + threadIdx.x;
    if (b >= BATCH) return;
    if (out_size >= 2 * BATCH) {       // complex64: (real, imag) interleaved
        out[2 * b]     = logamp[b];
        out[2 * b + 1] = 0.f;
    } else if (b < out_size) {
        out[b] = logamp[b];
    }
}

extern "C" void kernel_launch(void* const* d_in, const int* in_sizes, int n_in,
                              void* d_out, int out_size, void* d_ws, size_t ws_size,
                              hipStream_t stream)
{
    (void)in_sizes; (void)n_in; (void)ws_size;
    const int*   x    = (const int*)d_in[0];
    const float* Wemb = (const float*)d_in[1];
    const float* bemb = (const float*)d_in[2];
    const float* Wi   = (const float*)d_in[3];
    const float* bi   = (const float*)d_in[4];
    const float* Wh   = (const float*)d_in[5];
    const float* Wo   = (const float*)d_in[6];
    const float* bo   = (const float*)d_in[7];

    char*  ws  = (char*)d_ws;
    size_t off = 0;
    auto carve = [&](size_t bytes) -> char* {
        char* p = ws + off;
        off += bytes;
        off = (off + 255) & ~(size_t)255;
        return p;
    };
    const size_t wbytes = (size_t)3 * HDIM * HDIM * sizeof(__bf16);
    const size_t hf     = (size_t)BATCH * HDIM * sizeof(float);
    const size_t hb     = (size_t)BATCH * HDIM * sizeof(__bf16);

    __bf16* WiT0  = (__bf16*)carve(wbytes);
    __bf16* WiT1  = (__bf16*)carve(wbytes);
    __bf16* WhT1  = (__bf16*)carve(wbytes);
    float*  v0buf = (float*)carve(3 * HDIM * sizeof(float));
    float*  h0f[2] = { (float*)carve(hf), (float*)carve(hf) };
    float*  h1f[2] = { (float*)carve(hf), (float*)carve(hf) };
    __bf16* h0b[2] = { (__bf16*)carve(hb), (__bf16*)carve(hb) };
    __bf16* h1b[2] = { (__bf16*)carve(hb), (__bf16*)carve(hb) };
    float*  logamp = (float*)carve(BATCH * sizeof(float));

    // zero initial hidden states + log-amplitude accumulator (every call; deterministic)
    hipMemsetAsync(h0f[0], 0, hf, stream);
    hipMemsetAsync(h1f[0], 0, hf, stream);
    hipMemsetAsync(h0b[0], 0, hb, stream);
    hipMemsetAsync(h1b[0], 0, hb, stream);
    hipMemsetAsync(logamp, 0, BATCH * sizeof(float), stream);

    // weight prep: bf16 + transpose, and the rank-1 vector v0 = W_embed @ Wh[0]
    const int wblocks = (3 * HDIM * HDIM) / 256;
    transpose_w_kernel<<<wblocks, 256, 0, stream>>>(Wi, WiT0);
    transpose_w_kernel<<<wblocks, 256, 0, stream>>>(Wi + (size_t)HDIM * 3 * HDIM, WiT1);
    transpose_w_kernel<<<wblocks, 256, 0, stream>>>(Wh + (size_t)HDIM * 3 * HDIM, WhT1);
    v0_kernel<<<(3 * HDIM + 255) / 256, 256, 0, stream>>>(Wemb, Wh, v0buf);

    dim3 grid(HDIM / BN, BATCH / BM);   // (8, 8)
    for (int t = 0; t < SEQ; ++t) {
        const int p = t & 1, q = p ^ 1;
        // layer 0: gi = h0 @ Wi0 ; gh = xp*v0 (rank-1) ; carry = xp*Wemb + bemb
        gru_layer_kernel<false><<<grid, 256, 0, stream>>>(
            h0b[p], WiT0, nullptr, nullptr, bi, v0buf, x, t, Wemb, bemb, nullptr,
            h0f[q], h0b[q]);
        // layer 1: gi = h1 @ Wi1 ; gh = h0_new @ Wh1 ; carry = h0_new
        gru_layer_kernel<true><<<grid, 256, 0, stream>>>(
            h1b[p], WiT1, h0b[q], WhT1, bi + 3 * HDIM, nullptr, x, t, Wemb, bemb,
            h0f[q], h1f[q], h1b[q]);
        // output head + log-softmax accumulation
        head_kernel<<<BATCH / 256, 256, 0, stream>>>(h1f[q], Wo, bo, x, t, logamp);
    }
    finalize_kernel<<<BATCH / 256, 256, 0, stream>>>(logamp, (float*)d_out, out_size);
}
